// QWen3MoeAttention_18159121727913
// MI455X (gfx1250) — matrix operand order repaired and checked by reading, not compiled, not run on hardware
//
#include <hip/hip_runtime.h>

typedef __attribute__((ext_vector_type(16))) __bf16 v16bf;
typedef __attribute__((ext_vector_type(8)))  __bf16 v8bf;
typedef __attribute__((ext_vector_type(8)))  float  v8f;
typedef __attribute__((ext_vector_type(4)))  int    v4i;

typedef __attribute__((address_space(1))) v4i* gv4i_ptr;   // global 128b payload ptr
typedef __attribute__((address_space(3))) v4i* lv4i_ptr;   // LDS 128b payload ptr

#define HIDDEN 2048
#define NH     32
#define NKV    4
#define GQA    8
#define HD     128
#define TT     2048
#define NQ     (NH*HD)    // 4096
#define NKVD   (NKV*HD)   // 512

#if __has_builtin(__builtin_amdgcn_global_load_async_to_lds_b128) && \
    __has_builtin(__builtin_amdgcn_s_wait_asynccnt)
#define GEMM_ASYNC 1
#else
#define GEMM_ASYNC 0
#endif

// LDS panel row stride in bf16 elements (80 bytes) -> conflict-free b128 reads
#define PSTRIDE 40

// ---------------- elementwise f32 -> bf16 cast ----------------
__global__ __launch_bounds__(256) void cast_bf16_kernel(const float* __restrict__ x,
                                                        __bf16* __restrict__ y, int n) {
    int i = blockIdx.x * 256 + threadIdx.x;
    if (i < n) y[i] = (__bf16)x[i];
}

// ---------------- tiled transpose + cast: W[K][N] f32 -> WT[N][K] bf16 ----------------
__global__ __launch_bounds__(256) void transpose_cast_kernel(const float* __restrict__ W,
                                                             __bf16* __restrict__ WT,
                                                             int K, int N) {
    __shared__ float tile[32][33];
    int n0 = blockIdx.x * 32, k0 = blockIdx.y * 32;
    int tx = threadIdx.x, ty = threadIdx.y;          // 32 x 8
    #pragma unroll
    for (int i = 0; i < 32; i += 8)
        tile[ty + i][tx] = W[(size_t)(k0 + ty + i) * N + (n0 + tx)];
    __syncthreads();
    #pragma unroll
    for (int i = 0; i < 32; i += 8)
        WT[(size_t)(n0 + ty + i) * K + (k0 + tx)] = (__bf16)tile[tx][ty + i];
}

// ---------------- bf16 WMMA GEMM: C[M][N] f32 = A[M][K] * BT[N][K]^T ----------------
// 4-wave workgroup computes a 128x128 tile; each wave owns a 64x64 quadrant.
// Async path: double-buffered A/B K-panels (K-chunk = 32) staged in LDS via
// GLOBAL_LOAD_ASYNC_TO_LDS_B128 (ASYNCcnt), overlapping copy with WMMA.
__global__ __launch_bounds__(128) void gemm_bf16_kernel(const __bf16* __restrict__ A,
                                                        const __bf16* __restrict__ BT,
                                                        float* __restrict__ C,
                                                        int M, int N, int K) {
    const int tid   = threadIdx.x;                   // 0..127
    const int lane  = tid & 31;
    const int wave  = tid >> 5;                      // 0..3
    const int l16   = lane & 15;
    const int lhalf = lane >> 4;                     // 0 or 1
    const int mq = (wave >> 1) * 64;                 // quadrant offsets
    const int nq = (wave & 1) * 64;
    const int m0 = blockIdx.y * 128;
    const int n0 = blockIdx.x * 128;

    v8f acc[16];
    #pragma unroll
    for (int i = 0; i < 16; ++i) acc[i] = {};

#if GEMM_ASYNC
    __shared__ __bf16 pa[2][128 * PSTRIDE];          // A panel: 128 rows x 32 K (padded)
    __shared__ __bf16 pb[2][128 * PSTRIDE];          // B panel: 128 cols x 32 K (padded)

    // Each thread async-copies 4 x 16B of A and 4 x 16B of B per K-chunk (8 outstanding).
    auto issue = [&](int buf, int kk) {
        #pragma unroll
        for (int j = 0; j < 4; ++j) {
            int c = j * 128 + tid;                   // 0..511
            int row = c >> 2, seg = c & 3;
            const __bf16* ga = A + (size_t)(m0 + row) * K + kk + seg * 8;
            __bf16* la = &pa[buf][row * PSTRIDE + seg * 8];
            __builtin_amdgcn_global_load_async_to_lds_b128(
                (gv4i_ptr)(uintptr_t)ga,
                (lv4i_ptr)(unsigned)(uintptr_t)la, 0, 0);
        }
        #pragma unroll
        for (int j = 0; j < 4; ++j) {
            int c = j * 128 + tid;
            int col = c >> 2, seg = c & 3;
            const __bf16* gb = BT + (size_t)(n0 + col) * K + kk + seg * 8;
            __bf16* lb = &pb[buf][col * PSTRIDE + seg * 8];
            __builtin_amdgcn_global_load_async_to_lds_b128(
                (gv4i_ptr)(uintptr_t)gb,
                (lv4i_ptr)(unsigned)(uintptr_t)lb, 0, 0);
        }
    };

    issue(0, 0);
    const int nck = K >> 5;
    for (int ck = 0; ck < nck; ++ck) {
        const int cur  = ck & 1;
        const bool more = (ck + 1 < nck);
        if (more) issue(cur ^ 1, (ck + 1) << 5);
        // drain only the current panel's 8 async loads (next panel stays in flight)
        if (more) __builtin_amdgcn_s_wait_asynccnt(8);
        else      __builtin_amdgcn_s_wait_asynccnt(0);
        __syncthreads();

        v16bf a[4];
        #pragma unroll
        for (int mt = 0; mt < 4; ++mt) {
            const __bf16* ar = &pa[cur][(mq + mt * 16 + l16) * PSTRIDE + lhalf * 16];
            ((v8bf*)&a[mt])[0] = *(const v8bf*)(ar);
            ((v8bf*)&a[mt])[1] = *(const v8bf*)(ar + 8);
        }
        #pragma unroll
        for (int nt = 0; nt < 4; ++nt) {
            const __bf16* br = &pb[cur][(nq + nt * 16 + l16) * PSTRIDE + lhalf * 16];
            v16bf b;
            ((v8bf*)&b)[0] = *(const v8bf*)(br);
            ((v8bf*)&b)[1] = *(const v8bf*)(br + 8);
            #pragma unroll
            for (int mt = 0; mt < 4; ++mt)
                acc[mt * 4 + nt] = __builtin_amdgcn_wmma_f32_16x16x32_bf16(
                    false, a[mt], false, b, (short)0, acc[mt * 4 + nt], false, false);
        }
        __syncthreads();                             // all waves done before panel reuse
    }
#else
    // fallback: direct global feeds (per-wave 64x64 quadrant)
    for (int kk = 0; kk < K; kk += 32) {
        v16bf a[4];
        #pragma unroll
        for (int mt = 0; mt < 4; ++mt) {
            const __bf16* arow = A + (size_t)(m0 + mq + mt * 16 + l16) * K + kk;
            ((v8bf*)&a[mt])[0] = *(const v8bf*)(arow + lhalf * 16);
            ((v8bf*)&a[mt])[1] = *(const v8bf*)(arow + 8 + lhalf * 16);
        }
        #pragma unroll
        for (int nt = 0; nt < 4; ++nt) {
            const __bf16* brow = BT + (size_t)(n0 + nq + nt * 16 + l16) * K + kk + lhalf * 16;
            v16bf b = *(const v16bf*)brow;
            #pragma unroll
            for (int mt = 0; mt < 4; ++mt)
                acc[mt * 4 + nt] = __builtin_amdgcn_wmma_f32_16x16x32_bf16(
                    false, a[mt], false, b, (short)0, acc[mt * 4 + nt], false, false);
        }
    }
#endif

    #pragma unroll
    for (int mt = 0; mt < 4; ++mt)
        #pragma unroll
        for (int nt = 0; nt < 4; ++nt)
            #pragma unroll
            for (int v = 0; v < 8; ++v) {
                int row = m0 + mq + mt * 16 + v + lhalf * 8;  // C layout
                int col = n0 + nq + nt * 16 + l16;
                C[(size_t)row * N + col] = acc[mt * 4 + nt][v];
            }
}

// ---------------- fused per-head rmsnorm + RoPE, f32 in -> bf16 out ----------------
__global__ __launch_bounds__(128) void norm_rope_kernel(const float* __restrict__ x,
                                                        const float* __restrict__ w,
                                                        const int* __restrict__ pos,
                                                        __bf16* __restrict__ out,
                                                        int nheads) {
    __shared__ float sh[128];
    __shared__ float red[4];
    int tid = threadIdx.x;
    int bid = blockIdx.x;            // t * nheads + h
    int t = bid / nheads;
    size_t base = (size_t)bid * HD;
    float xv = x[base + tid];
    float ss = xv * xv;
    #pragma unroll
    for (int m = 1; m < 32; m <<= 1) ss += __shfl_xor(ss, m, 32);
    if ((tid & 31) == 0) red[tid >> 5] = ss;
    __syncthreads();
    float tot = red[0] + red[1] + red[2] + red[3];
    float r = rsqrtf(tot * (1.0f / HD) + 1e-6f);
    float xn = xv * r * w[tid];
    sh[tid] = xn;
    __syncthreads();
    float p = (float)pos[t];
    int i = tid & 63;
    float inv_freq = powf(10000.0f, -(float)(2 * i) / (float)HD);
    float fr = p * inv_freq;
    float c = cosf(fr), s = sinf(fr);
    float o = (tid < 64) ? (sh[i] * c - sh[i + 64] * s)
                         : (sh[i + 64] * c + sh[i] * s);
    out[base + tid] = (__bf16)o;
}

// ---------------- V f32 [T][NKV][HD] -> VT bf16 [NKV][HD][T] ----------------
__global__ __launch_bounds__(256) void vtrans_kernel(const float* __restrict__ vf,
                                                     __bf16* __restrict__ vt) {
    int idx = blockIdx.x * 256 + threadIdx.x;       // NKV*HD*T = 1048576
    int s    = idx & (TT - 1);
    int rest = idx >> 11;
    int d    = rest & (HD - 1);
    int hkv  = rest >> 7;
    vt[((size_t)hkv * HD + d) * TT + s] = (__bf16)vf[((size_t)s * NKV + hkv) * HD + d];
}

// ---------------- flash attention: 1 wave per (head, 16 query rows) ----------------
__global__ __launch_bounds__(32) void flash_attn_kernel(const __bf16* __restrict__ qh,
                                                        const __bf16* __restrict__ kh,
                                                        const __bf16* __restrict__ vt,
                                                        __bf16* __restrict__ attnb) {
    __shared__ __bf16 lp[16 * 32];                   // P tile, row-major [16][32]
    const int lane  = threadIdx.x;
    const int l16   = lane & 15;
    const int lhalf = lane >> 4;
    const int h  = blockIdx.y;
    const int t0 = blockIdx.x * 16;
    const int kv = h >> 3;                           // GQA = 8

    // Q tile (16 x 128) in A layout, loaded once: 4 K-chunks of 32
    v16bf qa[4];
    {
        const __bf16* qrow = qh + ((size_t)(t0 + l16) * NH + h) * HD;
        #pragma unroll
        for (int c = 0; c < 4; ++c) {
            int d0 = c * 32 + lhalf * 16;
            ((v8bf*)&qa[c])[0] = *(const v8bf*)(qrow + d0);
            ((v8bf*)&qa[c])[1] = *(const v8bf*)(qrow + d0 + 8);
        }
    }

    v8f o[8];
    #pragma unroll
    for (int i = 0; i < 8; ++i) o[i] = {};
    float m[8], l[8];
    #pragma unroll
    for (int v = 0; v < 8; ++v) { m[v] = -3.0e38f; l[v] = 0.0f; }

    const float scale = 0.08838834764831845f;        // 1/sqrt(128)
    const int s_end = t0 + 16;                       // keys with s <= t0+15

    for (int s0 = 0; s0 < s_end; s0 += 32) {
        // scores: two 16-column C tiles
        v8f sc[2];
        #pragma unroll
        for (int ct = 0; ct < 2; ++ct) {
            v8f acc = {};
            int scol = s0 + ct * 16 + l16;
            const __bf16* krow = kh + ((size_t)scol * NKV + kv) * HD;
            #pragma unroll
            for (int c = 0; c < 4; ++c) {
                v16bf b = *(const v16bf*)(krow + c * 32 + lhalf * 16);
                acc = __builtin_amdgcn_wmma_f32_16x16x32_bf16(
                    false, qa[c], false, b, (short)0, acc, false, false);
            }
            sc[ct] = acc;
        }
        // scale + causal mask (selects only; wave-uniform control flow)
        #pragma unroll
        for (int ct = 0; ct < 2; ++ct) {
            int s = s0 + ct * 16 + l16;
            #pragma unroll
            for (int v = 0; v < 8; ++v) {
                int t = t0 + v + lhalf * 8;
                float val = sc[ct][v] * scale;
                sc[ct][v] = (s <= t) ? val : -3.0e38f;
            }
        }
        // online softmax: row max (rows live in 16-lane halves -> xor 1,2,4,8)
        float alpha[8];
        #pragma unroll
        for (int v = 0; v < 8; ++v) {
            float vm = fmaxf(sc[0][v], sc[1][v]);
            vm = fmaxf(vm, __shfl_xor(vm, 1, 32));
            vm = fmaxf(vm, __shfl_xor(vm, 2, 32));
            vm = fmaxf(vm, __shfl_xor(vm, 4, 32));
            vm = fmaxf(vm, __shfl_xor(vm, 8, 32));
            float mn = fmaxf(m[v], vm);
            alpha[v] = expf(m[v] - mn);
            m[v] = mn;
        }
        #pragma unroll
        for (int ct = 0; ct < 2; ++ct)
            #pragma unroll
            for (int v = 0; v < 8; ++v)
                sc[ct][v] = expf(sc[ct][v] - m[v]);
        #pragma unroll
        for (int v = 0; v < 8; ++v) {
            float ps = sc[0][v] + sc[1][v];
            ps += __shfl_xor(ps, 1, 32);
            ps += __shfl_xor(ps, 2, 32);
            ps += __shfl_xor(ps, 4, 32);
            ps += __shfl_xor(ps, 8, 32);
            l[v] = l[v] * alpha[v] + ps;
        }
        #pragma unroll
        for (int nt = 0; nt < 8; ++nt)
            #pragma unroll
            for (int v = 0; v < 8; ++v)
                o[nt][v] *= alpha[v];

        // P: C layout f32 -> A layout bf16 via LDS
        __syncthreads();
        #pragma unroll
        for (int ct = 0; ct < 2; ++ct)
            #pragma unroll
            for (int v = 0; v < 8; ++v) {
                int r = v + lhalf * 8;
                lp[r * 32 + ct * 16 + l16] = (__bf16)sc[ct][v];
            }
        __syncthreads();
        v16bf paop;
        {
            const __bf16* prow = lp + l16 * 32 + lhalf * 16;
            ((v8bf*)&paop)[0] = *(const v8bf*)(prow);
            ((v8bf*)&paop)[1] = *(const v8bf*)(prow + 8);
        }
        // O += P @ V : B operand = VT[kv][d][s], contiguous over s -> one 32B load
        #pragma unroll
        for (int nt = 0; nt < 8; ++nt) {
            const __bf16* vrow =
                vt + ((size_t)kv * HD + nt * 16 + l16) * TT + s0 + lhalf * 16;
            v16bf b = *(const v16bf*)vrow;
            o[nt] = __builtin_amdgcn_wmma_f32_16x16x32_bf16(
                false, paop, false, b, (short)0, o[nt], false, false);
        }
    }

    // epilogue: divide by row sums, store bf16 attn[t][h*HD + d]
    #pragma unroll
    for (int nt = 0; nt < 8; ++nt)
        #pragma unroll
        for (int v = 0; v < 8; ++v) {
            int t = t0 + v + lhalf * 8;
            int d = nt * 16 + l16;
            attnb[(size_t)t * NQ + h * HD + d] = (__bf16)(o[nt][v] / l[v]);
        }
}

extern "C" void kernel_launch(void* const* d_in, const int* in_sizes, int n_in,
                              void* d_out, int out_size, void* d_ws, size_t ws_size,
                              hipStream_t stream) {
    const int*   positions = (const int*)d_in[0];
    const float* hs        = (const float*)d_in[1];
    const float* Wq        = (const float*)d_in[2];
    const float* Wk        = (const float*)d_in[3];
    const float* Wv        = (const float*)d_in[4];
    const float* Wo        = (const float*)d_in[5];
    const float* qw        = (const float*)d_in[6];
    const float* kw        = (const float*)d_in[7];
    float* out = (float*)d_out;

    char* p = (char*)d_ws;
    auto alloc = [&](size_t bytes) -> char* {
        char* r = p;
        p += (bytes + 255) & ~(size_t)255;
        return r;
    };
    __bf16* hsb  = (__bf16*)alloc((size_t)TT * HIDDEN * 2);
    __bf16* WqT  = (__bf16*)alloc((size_t)NQ * HIDDEN * 2);
    __bf16* WkT  = (__bf16*)alloc((size_t)NKVD * HIDDEN * 2);
    __bf16* WvT  = (__bf16*)alloc((size_t)NKVD * HIDDEN * 2);
    __bf16* WoT  = (__bf16*)alloc((size_t)HIDDEN * NQ * 2);
    float*  qf   = (float*) alloc((size_t)TT * NQ * 4);
    float*  kf   = (float*) alloc((size_t)TT * NKVD * 4);
    float*  vf   = (float*) alloc((size_t)TT * NKVD * 4);
    __bf16* qhat = (__bf16*)alloc((size_t)TT * NQ * 2);
    __bf16* khat = (__bf16*)alloc((size_t)TT * NKVD * 2);
    __bf16* vtb  = (__bf16*)alloc((size_t)NKV * HD * TT * 2);
    __bf16* attnb = (__bf16*)qf;   // overlay: qf is dead after norm_rope(q)

    // 1) casts / transposes
    cast_bf16_kernel<<<(TT * HIDDEN) / 256, 256, 0, stream>>>(hs, hsb, TT * HIDDEN);
    transpose_cast_kernel<<<dim3(NQ / 32, HIDDEN / 32), dim3(32, 8), 0, stream>>>(Wq, WqT, HIDDEN, NQ);
    transpose_cast_kernel<<<dim3(NKVD / 32, HIDDEN / 32), dim3(32, 8), 0, stream>>>(Wk, WkT, HIDDEN, NKVD);
    transpose_cast_kernel<<<dim3(NKVD / 32, HIDDEN / 32), dim3(32, 8), 0, stream>>>(Wv, WvT, HIDDEN, NKVD);
    transpose_cast_kernel<<<dim3(HIDDEN / 32, NQ / 32), dim3(32, 8), 0, stream>>>(Wo, WoT, NQ, HIDDEN);

    // 2) QKV projections (bf16 WMMA, f32 accumulate, async LDS double-buffered)
    gemm_bf16_kernel<<<dim3(NQ / 128, TT / 128), 128, 0, stream>>>(hsb, WqT, qf, TT, NQ, HIDDEN);
    gemm_bf16_kernel<<<dim3(NKVD / 128, TT / 128), 128, 0, stream>>>(hsb, WkT, kf, TT, NKVD, HIDDEN);
    gemm_bf16_kernel<<<dim3(NKVD / 128, TT / 128), 128, 0, stream>>>(hsb, WvT, vf, TT, NKVD, HIDDEN);

    // 3) rmsnorm + RoPE -> bf16
    norm_rope_kernel<<<TT * NH, 128, 0, stream>>>(qf, qw, positions, qhat, NH);
    norm_rope_kernel<<<TT * NKV, 128, 0, stream>>>(kf, kw, positions, khat, NKV);

    // 4) V transpose -> [kv][d][T] bf16
    vtrans_kernel<<<(NKV * HD * TT) / 256, 256, 0, stream>>>(vf, vtb);

    // 5) flash attention
    flash_attn_kernel<<<dim3(TT / 16, NH), 32, 0, stream>>>(qhat, khat, vtb, attnb);

    // 6) output projection -> f32 d_out
    gemm_bf16_kernel<<<dim3(HIDDEN / 128, TT / 128), 128, 0, stream>>>(attnb, WoT, out, TT, HIDDEN, NQ);
}
